// VectorQuantize_19310172963581
// MI455X (gfx1250) — matrix operand order repaired and checked by reading, not compiled, not run on hardware
//
#include <hip/hip_runtime.h>
#include <hip/hip_bf16.h>

typedef _Float16 v16h __attribute__((ext_vector_type(16)));
typedef float    v8f  __attribute__((ext_vector_type(8)));
typedef unsigned int u32x4 __attribute__((ext_vector_type(4)));
typedef unsigned int u32x8 __attribute__((ext_vector_type(8)));

#define E_DIM    64
#define N_EMBED  1024
#define N_TOKENS 262144

// Workspace layout (bpack+nrm contiguous so ONE TDM op stages both into LDS):
//   [0, 128K)        bpack  (65536 f16)  -> LDS offset 0
//   [128K, 132K)     nrm    (1024 f32)   -> LDS offset 131072
//   [132K, 388K)     embedT (65536 f32)  (stays in L2, gather source)
#define WS_BPACK   0
#define WS_NRM     131072
#define WS_EMBEDT  135168
#define STAGE_BYTES 135168        // 33792 dwords, DMA'd to LDS per block

// ---------------------------------------------------------------------------
// Prep kernel: one thread per code j (1024 total).
//  - bpack: f16 B fragments in exact WMMA lane layout:
//      entry ((t*2+kb)*32 + lane) holds 16 halfs:
//        lane<16 : K = kb*32 + {0..7,16..23}   of code N = 16t+lane
//        lane>=16: K = kb*32 + {8..15,24..31}  of code N = 16t+lane-16
//  - nrm_half[j]  = 0.5*||e_j||^2
//  - embedT[j][d] = embed[d][j]   (contiguous rows for the final gather)
//  - out_embed    = copy of embed to d_out tail
// ---------------------------------------------------------------------------
__global__ __launch_bounds__(256) void vq_prep(
    const float* __restrict__ embed,
    float* __restrict__ out_embed,
    float* __restrict__ nrm_half,
    float* __restrict__ embedT,
    _Float16* __restrict__ bpack)
{
    const int j = blockIdx.x * blockDim.x + threadIdx.x;   // code id 0..1023
    float nrm = 0.0f;
    _Float16 h[64];
#pragma unroll
    for (int d = 0; d < 64; ++d) {
        float v = embed[d * N_EMBED + j];    // coalesced across block per d
        nrm += v * v;
        embedT[j * 64 + d] = v;
        out_embed[d * N_EMBED + j] = v;
        h[d] = (_Float16)v;
    }
    nrm_half[j] = 0.5f * nrm;

    const int t = j >> 4;
    const int l = j & 15;
#pragma unroll
    for (int kb = 0; kb < 2; ++kb) {
#pragma unroll
        for (int half = 0; half < 2; ++half) {
            const int lane = half * 16 + l;
            const int base = ((t * 2 + kb) * 32 + lane) * 16;
#pragma unroll
            for (int k = 0; k < 16; ++k)
                bpack[base + k] = h[kb * 32 + (k < 8 ? half * 8 + k : 16 + half * 8 + (k - 8))];
        }
    }
}

// ---------------------------------------------------------------------------
// Main kernel: 256 threads = 8 wave32; each wave owns 16 tokens.
// Wave 0 issues one tensor_load_to_lds DMA of the 132 KB bpack+bias region
// (L2-resident) into LDS; TENSORcnt + workgroup barrier publish it.
// ---------------------------------------------------------------------------
__global__ __launch_bounds__(256) void vq_main(
    const float* __restrict__ x,
    const void* __restrict__ stage_src,      // ws base: bpack+nrm, 132 KB
    const float* __restrict__ embedT,
    float* __restrict__ qout,
    float* __restrict__ in_copy,
    float* __restrict__ qig,
    float* __restrict__ idx_out)
{
    extern __shared__ char smem[];
    _Float16* sB = (_Float16*)smem;                 // 65536 halfs = 128 KB
    float*    sN = (float*)(smem + WS_NRM);         // 1024 f32   =   4 KB

    const int tid  = threadIdx.x;
    const int lane = tid & 31;
    const int wave = tid >> 5;
    const int tok0 = (blockIdx.x * 8 + wave) * 16;  // this wave's 16 tokens
    const float* xrow = x + (size_t)tok0 * 64;

    // ---- Tensor Data Mover: DMA [stage_src, +132KB) -> LDS [0, 132KB) ----
    if (wave == 0) {
        const unsigned long long ga = (unsigned long long)stage_src;
        const unsigned int ldsOff = (unsigned int)(uintptr_t)smem;  // LDS byte offset

        u32x4 g0;
        g0[0] = 1u;                                   // count=1, user descriptor
        g0[1] = ldsOff;                               // lds_addr (bytes)
        g0[2] = (unsigned int)(ga & 0xFFFFFFFFu);     // global_addr[31:0]
        g0[3] = (unsigned int)((ga >> 32) & 0x01FFFFFFu) | 0x80000000u; // [56:32] | type=2

        const unsigned int nd = STAGE_BYTES / 4;      // 33792 dwords
        u32x8 g1;
        g1[0] = 2u << 16;                             // data_size=4B; no mask/pad/iter
        g1[1] = (nd & 0xFFFFu) << 16;                 // tensor_dim0[15:0]
        g1[2] = (nd >> 16) | (1u << 16);              // tensor_dim0[31:16] | tensor_dim1=1
        g1[3] = (nd & 0xFFFFu) << 16;                 // tile_dim0 = nd
        g1[4] = 1u;                                   // tile_dim1=1, tile_dim2=0
        g1[5] = nd;                                   // tensor_dim0_stride[31:0]
        g1[6] = 0u;                                   // stride0 hi / stride1 lo
        g1[7] = 0u;                                   // tensor_dim1_stride hi

        asm volatile("tensor_load_to_lds %0, %1"
                     :
                     : "s"(g0), "s"(g1)
                     : "memory");
        __builtin_amdgcn_s_wait_tensorcnt(0);         // s_wait_tensorcnt 0
    }

    // Prefetch this wave's 4 KB input tile while the DMA runs.
    __builtin_prefetch(xrow + lane * 32, 0, 0);       // global_prefetch_b8

    __syncthreads();                                  // publish LDS to all waves

    // ---- load A tile (16 tokens x 64 dims), fuse the "input" output copy ----
    // 16-bit A 16x32 layout: lanes 0-15 -> K blocks {0..7,16..23} of row=lane,
    //                        lanes 16-31 -> K blocks {8..15,24..31} of row=lane-16.
    const int row = lane & 15;
    const int kb0 = (lane & 16) ? 8 : 0;
    const float4* src = (const float4*)(xrow + row * 64);
    float4* dst = (float4*)(in_copy + (size_t)(tok0 + row) * 64);

    float fl[32];
#pragma unroll
    for (int g = 0; g < 4; ++g) {
        const int off = kb0 + g * 16;               // float offset, 16B aligned
        float4 u = src[off / 4];
        float4 v = src[off / 4 + 1];
        dst[off / 4]     = u;                       // input copy (exact coverage)
        dst[off / 4 + 1] = v;
        fl[8 * g + 0] = u.x; fl[8 * g + 1] = u.y; fl[8 * g + 2] = u.z; fl[8 * g + 3] = u.w;
        fl[8 * g + 4] = v.x; fl[8 * g + 5] = v.y; fl[8 * g + 6] = v.z; fl[8 * g + 7] = v.w;
    }
    v16h a_lo, a_hi;                                // K=0..31 and K=32..63 fragments
#pragma unroll
    for (int e = 0; e < 16; ++e) {
        a_lo[e] = (_Float16)fl[e];
        a_hi[e] = (_Float16)fl[16 + e];
    }

    // ---- sweep 64 code tiles: score = x.e - 0.5*||e||^2, track argmax ----
    float best[8];
    int   bidx[8];
#pragma unroll
    for (int r = 0; r < 8; ++r) { best[r] = -3.4e38f; bidx[r] = 0; }
    const int nsel = lane & 15;

#pragma unroll 4
    for (int t = 0; t < 64; ++t) {
        const v16h b_lo = *(const v16h*)(sB + t * 1024 + lane * 16);
        const v16h b_hi = *(const v16h*)(sB + t * 1024 + 512 + lane * 16);
        v8f c = {};
        c = __builtin_amdgcn_wmma_f32_16x16x32_f16(false, a_lo, false, b_lo,
                                                   (short)0, c, false, false);
        c = __builtin_amdgcn_wmma_f32_16x16x32_f16(false, a_hi, false, b_hi,
                                                   (short)0, c, false, false);
        const float bias = sN[t * 16 + nsel];
        const int   code = t * 16 + nsel;
#pragma unroll
        for (int r = 0; r < 8; ++r) {
            const float s = c[r] - bias;
            if (s > best[r]) { best[r] = s; bidx[r] = code; }  // ascending codes => first-min
        }
    }

    // ---- argmax reduction across the 16 lanes of each half-wave ----
    // C layout: lanes 0-15 hold tokens 0..7 (VGPR r -> token r),
    //           lanes 16-31 hold tokens 8..15.
#pragma unroll
    for (int r = 0; r < 8; ++r) {
#pragma unroll
        for (int m = 1; m < 16; m <<= 1) {
            const float ob = __shfl_xor(best[r], m, 32);
            const int   oi = __shfl_xor(bidx[r], m, 32);
            if (ob > best[r] || (ob == best[r] && oi < bidx[r])) {
                best[r] = ob; bidx[r] = oi;         // lowest index on ties
            }
        }
    }

    // ---- gather winning codes (coalesced from embedT) + quantize/grad/idx ----
#pragma unroll
    for (int m = 0; m < 16; ++m) {
        const int j = (m < 8) ? __shfl(bidx[m & 7], 0, 32)
                              : __shfl(bidx[m & 7], 16, 32);
        if (lane == 0) idx_out[tok0 + m] = (float)j;
        const float2 v = ((const float2*)(embedT + (size_t)j * 64))[lane];
        ((float2*)(qout + (size_t)(tok0 + m) * 64))[lane] = v;
        ((float2*)(qig  + (size_t)(tok0 + m) * 64))[lane] = v;   // STE fwd == quantize
    }
}

// ---------------------------------------------------------------------------
extern "C" void kernel_launch(void* const* d_in, const int* in_sizes, int n_in,
                              void* d_out, int out_size, void* d_ws, size_t ws_size,
                              hipStream_t stream)
{
    const float* x     = (const float*)d_in[0];   // [262144, 64]
    const float* embed = (const float*)d_in[1];   // [64, 1024]

    float* out      = (float*)d_out;              // outputs concatenated flat
    float* qout     = out;                                            // quantize
    float* in_copy  = out + (size_t)N_TOKENS * E_DIM;                 // input
    float* qig      = out + 2 * (size_t)N_TOKENS * E_DIM;             // STE grad path
    float* idx_out  = out + 3 * (size_t)N_TOKENS * E_DIM;             // embed_idxs
    float* out_embed = idx_out + N_TOKENS;                            // embed copy

    char* ws = (char*)d_ws;
    _Float16* bpack  = (_Float16*)(ws + WS_BPACK);   // 128 KB
    float*    nrm    = (float*)(ws + WS_NRM);        //   4 KB
    float*    embedT = (float*)(ws + WS_EMBEDT);     // 256 KB

    vq_prep<<<4, 256, 0, stream>>>(embed, out_embed, nrm, embedT, bpack);

    const size_t ldsBytes = STAGE_BYTES;             // 132 KB of 320 KB/WGP
    vq_main<<<N_TOKENS / 128, 256, ldsBytes, stream>>>(
        x, ws, embedT, qout, in_copy, qig, idx_out);
}
